// TransformerBrain_50663434223764
// MI455X (gfx1250) — matrix operand order repaired and checked by reading, not compiled, not run on hardware
//
#include <hip/hip_runtime.h>
#include <math.h>
#include <stdint.h>

typedef __attribute__((ext_vector_type(16))) _Float16 v16h;
typedef __attribute__((ext_vector_type(8)))  float    v8f;
typedef __attribute__((ext_vector_type(4)))  unsigned int v4u;
typedef __attribute__((ext_vector_type(8)))  int      v8i;
typedef __attribute__((ext_vector_type(4)))  int      v4i;

#define RAYS 64
#define RAYF 8
#define RICHN 32
#define EDIM 32
#define HIDN 128
#define OBSN 544
#define WPB  4   // waves (samples) per workgroup

#if __has_builtin(__builtin_amdgcn_tensor_load_to_lds) && \
    __has_builtin(__builtin_amdgcn_s_wait_tensorcnt)
#define USE_TDM 1
#else
#define USE_TDM 0
#endif

struct Params {
  const float *obs, *ray_ln_g, *ray_ln_b, *ray_W, *ray_b;
  const float *rich_ln_g, *rich_ln_b, *rich_W, *rich_b, *pos;
  const float *ca_Wq,*ca_bq,*ca_Wk,*ca_bk,*ca_Wv,*ca_bv,*ca_Wo,*ca_bo;
  const float *ca_ln1_g,*ca_ln1_b,*ca_W1,*ca_b1,*ca_W2,*ca_b2,*ca_ln2_g,*ca_ln2_b;
  const float *sa_Wq,*sa_bq,*sa_Wk,*sa_bk,*sa_Wv,*sa_bv,*sa_Wo,*sa_bo;
  const float *sa_ln1_g,*sa_ln1_b,*sa_W1,*sa_b1,*sa_W2,*sa_b2,*sa_ln2_g,*sa_ln2_b;
  const float *fc_in_W,*fc_in_b,*fc1_W,*fc1_b,*actor_W,*actor_b,*critic_W,*critic_b;
  float* out; int nB;
};

__device__ __forceinline__ float gelu_exact(float x) {
  return 0.5f * x * (1.0f + erff(x * 0.7071067811865475f));
}

__device__ __forceinline__ v8f wmma32(v16h a, v16h b, v8f c) {
  // v_wmma_f32_16x16x32_f16 : D = A(16x32 f16) x B(32x16 f16) + C(f32)
  return __builtin_amdgcn_wmma_f32_16x16x32_f16(false, a, false, b, (short)0, c,
                                                false, false);
}

// ---- fragment loaders (contiguous k = 16*(lane/16) + e in A and B, wave32) -------------------------
__device__ __forceinline__ v16h load_a(const _Float16* src, int r0, int stride, int k0) {
  int lane = threadIdx.x & 31;
  int m = lane & 15, h = lane >> 4;
  const _Float16* rowp = src + (r0 + m) * stride + k0 + h * 16;
  v16h a;
#pragma unroll
  for (int e = 0; e < 8; ++e) a[e] = rowp[e];
#pragma unroll
  for (int e = 8; e < 16; ++e) a[e] = rowp[e];
  return a;
}

__device__ __forceinline__ v16h load_b(const _Float16* wt, int n0, int K, int k0) {
  int lane = threadIdx.x & 31;
  int n = lane & 15, h = lane >> 4;
  const _Float16* rowp = wt + (n0 + n) * K + k0 + h * 16;
  v16h b;
#pragma unroll
  for (int e = 0; e < 16; ++e) b[e] = rowp[e];
  return b;
}

// ---- C-fragment stores: C f32 16x16, VGPR r -> row r + 8*(lane>=16), col lane&15
__device__ __forceinline__ void st_f16(_Float16* dst, int r0, int c0, int stride,
                                       v8f c, const float* bias, float scale) {
  int lane = threadIdx.x & 31, n = lane & 15, h = lane >> 4;
  float bb = bias ? bias[c0 + n] : 0.0f;
#pragma unroll
  for (int r = 0; r < 8; ++r)
    dst[(r0 + r + h * 8) * stride + c0 + n] = (_Float16)((c[r] + bb) * scale);
}

__device__ __forceinline__ void st_f16_gelu(_Float16* dst, int r0, int c0, int stride,
                                            v8f c, const float* bias) {
  int lane = threadIdx.x & 31, n = lane & 15, h = lane >> 4;
  float bb = bias[c0 + n];
#pragma unroll
  for (int r = 0; r < 8; ++r)
    dst[(r0 + r + h * 8) * stride + c0 + n] = (_Float16)gelu_exact(c[r] + bb);
}

__device__ __forceinline__ void st_f16_T(_Float16* dst, int r0, int c0, int strideT,
                                         v8f c, const float* bias) {
  int lane = threadIdx.x & 31, n = lane & 15, h = lane >> 4;
  float bb = bias[c0 + n];
#pragma unroll
  for (int r = 0; r < 8; ++r)
    dst[(c0 + n) * strideT + (r0 + r + h * 8)] = (_Float16)(c[r] + bb);
}

__device__ __forceinline__ void acc_f32(float* dst, int r0, int c0, int stride,
                                        v8f c, const float* bias) {
  int lane = threadIdx.x & 31, n = lane & 15, h = lane >> 4;
  float bb = bias[c0 + n];
#pragma unroll
  for (int r = 0; r < 8; ++r)
    dst[(r0 + r + h * 8) * stride + c0 + n] += c[r] + bb;
}

// ---- misc helpers ----------------------------------------------------------
__device__ __forceinline__ float wsum32(float v) {
#pragma unroll
  for (int o = 16; o > 0; o >>= 1) v += __shfl_xor(v, o, 32);
  return v;
}

__device__ __forceinline__ void ln_tokens(float* x, _Float16* xf,
                                          const float* g, const float* b, int lane) {
#pragma unroll
  for (int ti = 0; ti < 2; ++ti) {
    int t = lane * 2 + ti;
    float* row = x + t * EDIM;
    float m = 0.f;
    for (int e = 0; e < EDIM; ++e) m += row[e];
    m *= (1.0f / EDIM);
    float v = 0.f;
    for (int e = 0; e < EDIM; ++e) { float d = row[e] - m; v += d * d; }
    v *= (1.0f / EDIM);
    float rs = rsqrtf(v + 1e-5f);
    for (int e = 0; e < EDIM; ++e) {
      float o = (row[e] - m) * rs * g[e] + b[e];
      row[e] = o;
      xf[t * EDIM + e] = (_Float16)o;
    }
  }
}

__device__ __forceinline__ void ffn_block(float* x, _Float16* xf, _Float16* tmp,
                                          const _Float16* W1, const float* b1,
                                          const _Float16* W2, const float* b2) {
#pragma unroll
  for (int rt = 0; rt < 4; ++rt)
#pragma unroll
    for (int ct = 0; ct < 4; ++ct) {
      v8f c = {};
      c = wmma32(load_a(xf, rt * 16, EDIM, 0), load_b(W1, ct * 16, EDIM, 0), c);
      st_f16_gelu(tmp, rt * 16, ct * 16, 64, c, b1);
    }
#pragma unroll
  for (int rt = 0; rt < 4; ++rt)
#pragma unroll
    for (int ct = 0; ct < 2; ++ct) {
      v8f c = {};
#pragma unroll
      for (int j = 0; j < 2; ++j)
        c = wmma32(load_a(tmp, rt * 16, 64, j * 32), load_b(W2, ct * 16, 64, j * 32), c);
      acc_f32(x, rt * 16, ct * 16, EDIM, c, b2);
    }
}

// ---- kernel ----------------------------------------------------------------
__global__ __launch_bounds__(128, 1)
void brain_fused_kernel(Params p) {
  // f16 weights staged in [N][K] layout (B-fragment friendly)
  __shared__ _Float16 w_caW1[64 * 32];
  __shared__ _Float16 w_caW2[32 * 64];
  __shared__ _Float16 w_saWq[32 * 32], w_saWk[32 * 32], w_saWv[32 * 32], w_saWo[32 * 32];
  __shared__ _Float16 w_saW1[64 * 32];
  __shared__ _Float16 w_saW2[32 * 64];
  // TDM-staged observations for this workgroup's 4 samples
  __shared__ __align__(16) float lobs[WPB][OBSN];
  // per-wave (per-sample) activation scratch
  __shared__ float     sx[WPB][RAYS * EDIM];
  __shared__ _Float16  sxf[WPB][RAYS * EDIM];
  __shared__ _Float16  stmp[WPB][RAYS * 64];
  __shared__ _Float16  sk16[WPB][RAYS * EDIM];
  __shared__ _Float16  svT16[WPB][EDIM * RAYS];
  __shared__ float     s_rich[WPB][EDIM];
  __shared__ float     s_vec[WPB][EDIM];
  __shared__ float     s_vca[WPB][EDIM];
  __shared__ float     s_h1[WPB][HIDN], s_h2[WPB][HIDN];

  // ---- issue TDM load of obs tile (4 rows x 544 f32) into LDS ----
#if USE_TDM
  if (threadIdx.x < 32) {
    int row0 = blockIdx.x * WPB;
    int rows = p.nB - row0; if (rows > WPB) rows = WPB;
    uint64_t gaddr = (uint64_t)(uintptr_t)(p.obs + (size_t)row0 * OBSN);
    uint32_t laddr = (uint32_t)(uintptr_t)(&lobs[0][0]);  // flat LDS addr[31:0] = LDS byte offset
    // D# group0: count=1, lds_addr, global_addr[56:0], type=2 ("image")
    v4u g0;
    g0[0] = 1u;
    g0[1] = laddr;
    g0[2] = (uint32_t)(gaddr & 0xFFFFFFFFu);
    g0[3] = (uint32_t)((gaddr >> 32) & 0x01FFFFFFu) | (2u << 30);
    // D# group1: data_size=4B (code 2), tensor 544 x rows, tile 544 x WPB, stride0=544
    v8i g1;
    g1[0] = (int)(2u << 16);                 // data_size = 2 -> 4 bytes
    g1[1] = (int)(((uint32_t)OBSN) << 16);   // tensor_dim0[15:0] in bits 63:48
    g1[2] = (int)(((uint32_t)rows) << 16);   // tensor_dim1[15:0] in bits 95:80
    g1[3] = (int)(((uint32_t)OBSN) << 16);   // tile_dim0 in bits 127:112
    g1[4] = WPB;                             // tile_dim1 in bits 143:128
    g1[5] = OBSN;                            // tensor_dim0_stride low 32 (bits 191:160)
    g1[6] = 0;
    g1[7] = 0;
    v4i g2 = {}; v4i g3 = {};
#if __has_include(<hip/amd_detail/amd_gfx1250_TDM.h>)
    v8i g4 = {};
    __builtin_amdgcn_tensor_load_to_lds(g0, g1, g2, g3, g4, 0);   // 6-arg toolchain
#else
    __builtin_amdgcn_tensor_load_to_lds(g0, g1, g2, g3, 0);       // 5-arg toolchain
#endif
    __builtin_amdgcn_s_wait_tensorcnt(0);
  }
#else
  {
    int row0 = blockIdx.x * WPB;
    int total = WPB * OBSN;
    const float* src = p.obs + (size_t)row0 * OBSN;
    int avail = (p.nB - row0) * OBSN; if (avail > total) avail = total;
    for (int i = threadIdx.x; i < total; i += blockDim.x)
      lobs[0][i] = (i < avail) ? src[i] : 0.0f;
  }
#endif

  // ---- cooperative weight staging (transpose to [N][K], f32 -> f16) ----
  {
    const float* srcs[8] = { p.ca_W1, p.ca_W2, p.sa_Wq, p.sa_Wk, p.sa_Wv, p.sa_Wo, p.sa_W1, p.sa_W2 };
    _Float16*    dsts[8] = { w_caW1,  w_caW2,  w_saWq,  w_saWk,  w_saWv,  w_saWo,  w_saW1,  w_saW2 };
    const int    Ks[8]   = { 32, 64, 32, 32, 32, 32, 32, 64 };
    const int    Ns[8]   = { 64, 32, 32, 32, 32, 32, 64, 32 };
    for (int w = 0; w < 8; ++w) {
      int K = Ks[w], N = Ns[w];
      for (int i = threadIdx.x; i < K * N; i += blockDim.x) {
        int n = i / K, k = i % K;
        dsts[w][i] = (_Float16)srcs[w][k * N + n];
      }
    }
  }
  __syncthreads();

  const int wave = threadIdx.x >> 5;
  const int lane = threadIdx.x & 31;
  const int s = blockIdx.x * WPB + wave;
  if (s >= p.nB) return;  // wave-uniform exit; EXEC stays all-ones for live waves

  float*    x   = sx[wave];
  _Float16* xf  = sxf[wave];
  _Float16* tmp = stmp[wave];
  _Float16* k16 = sk16[wave];
  _Float16* vT  = svT16[wave];
  const float* obs = lobs[wave];

  // ---- ray tokens: LN(8) @ ray_W + ray_b + pos (VALU; K=8 too small for WMMA) ----
#pragma unroll
  for (int ti = 0; ti < 2; ++ti) {
    int t = lane * 2 + ti;
    float r[RAYF], y[RAYF];
    float m = 0.f;
#pragma unroll
    for (int f = 0; f < RAYF; ++f) { r[f] = obs[t * RAYF + f]; m += r[f]; }
    m *= (1.0f / RAYF);
    float var = 0.f;
#pragma unroll
    for (int f = 0; f < RAYF; ++f) { float d = r[f] - m; var += d * d; }
    var *= (1.0f / RAYF);
    float rs = rsqrtf(var + 1e-5f);
#pragma unroll
    for (int f = 0; f < RAYF; ++f) y[f] = (r[f] - m) * rs * p.ray_ln_g[f] + p.ray_ln_b[f];
    for (int e = 0; e < EDIM; ++e) {
      float acc = p.ray_b[e] + p.pos[t * EDIM + e];
#pragma unroll
      for (int f = 0; f < RAYF; ++f) acc += y[f] * p.ray_W[f * EDIM + e];
      x[t * EDIM + e] = acc;
      xf[t * EDIM + e] = (_Float16)acc;
    }
  }

  // ---- rich token: LN(32) @ rich_W + rich_b ----
  {
    float rv = obs[RAYS * RAYF + lane];
    float m = wsum32(rv) * (1.0f / RICHN);
    float d = rv - m;
    float var = wsum32(d * d) * (1.0f / RICHN);
    float ln = d * rsqrtf(var + 1e-5f) * p.rich_ln_g[lane] + p.rich_ln_b[lane];
    s_vec[wave][lane] = ln;
    float rt = p.rich_b[lane];
    for (int k = 0; k < RICHN; ++k) rt += s_vec[wave][k] * p.rich_W[k * EDIM + lane];
    s_rich[wave][lane] = rt;
  }

  // ---- cross-attention: softmax over 1 key == 1 exactly -> out = v@Wo+bo ----
  {
    float vv = p.ca_bv[lane];
    for (int k = 0; k < EDIM; ++k) vv += s_rich[wave][k] * p.ca_Wv[k * EDIM + lane];
    s_vca[wave][lane] = vv;
    float av = p.ca_bo[lane];
    for (int k = 0; k < EDIM; ++k) av += s_vca[wave][k] * p.ca_Wo[k * EDIM + lane];
    s_vec[wave][lane] = av;
  }
#pragma unroll
  for (int ti = 0; ti < 2; ++ti) {
    int t = lane * 2 + ti;
    for (int e = 0; e < EDIM; ++e) x[t * EDIM + e] += s_vec[wave][e];
  }
  ln_tokens(x, xf, p.ca_ln1_g, p.ca_ln1_b, lane);
  ffn_block(x, xf, tmp, w_caW1, p.ca_b1, w_caW2, p.ca_b2);
  ln_tokens(x, xf, p.ca_ln2_g, p.ca_ln2_b, lane);

  // ---- self-attention (WMMA throughout) ----
  const float scale = 0.17677669529663687f;  // 1/sqrt(32)
#pragma unroll
  for (int rt = 0; rt < 4; ++rt)
#pragma unroll
    for (int ct = 0; ct < 2; ++ct) {
      v8f c = {};
      c = wmma32(load_a(xf, rt * 16, EDIM, 0), load_b(w_saWk, ct * 16, EDIM, 0), c);
      st_f16(k16, rt * 16, ct * 16, EDIM, c, p.sa_bk, 1.0f);
    }
#pragma unroll
  for (int rt = 0; rt < 4; ++rt)
#pragma unroll
    for (int ct = 0; ct < 2; ++ct) {
      v8f c = {};
      c = wmma32(load_a(xf, rt * 16, EDIM, 0), load_b(w_saWv, ct * 16, EDIM, 0), c);
      st_f16_T(vT, rt * 16, ct * 16, RAYS, c, p.sa_bv);
    }
#pragma unroll
  for (int rt = 0; rt < 4; ++rt) {
    v8f c0 = {}, c1 = {};
    v16h a = load_a(xf, rt * 16, EDIM, 0);
    c0 = wmma32(a, load_b(w_saWq, 0, EDIM, 0), c0);
    c1 = wmma32(a, load_b(w_saWq, 16, EDIM, 0), c1);
    st_f16(xf, rt * 16, 0, EDIM, c0, p.sa_bq, scale);
    st_f16(xf, rt * 16, 16, EDIM, c1, p.sa_bq, scale);
  }
#pragma unroll
  for (int rt = 0; rt < 4; ++rt)
#pragma unroll
    for (int ct = 0; ct < 4; ++ct) {
      v8f c = {};
      c = wmma32(load_a(xf, rt * 16, EDIM, 0), load_b(k16, ct * 16, EDIM, 0), c);
      st_f16(tmp, rt * 16, ct * 16, 64, c, nullptr, 1.0f);
    }
  // row softmax over 64 keys (lane handles rows lane and lane+32)
#pragma unroll
  for (int ri = 0; ri < 2; ++ri) {
    int r = lane + ri * 32;
    _Float16* row = tmp + r * 64;
    float mx = -3.0e38f;
    for (int j = 0; j < 64; ++j) { float v = (float)row[j]; mx = v > mx ? v : mx; }
    float sm = 0.f;
    for (int j = 0; j < 64; ++j) { float e = __expf((float)row[j] - mx); sm += e; row[j] = (_Float16)e; }
    float inv = 1.0f / sm;
    for (int j = 0; j < 64; ++j) row[j] = (_Float16)((float)row[j] * inv);
  }
  // attn = P(64x64) @ V -> packed f16 into tmp cols [0,32)
#pragma unroll
  for (int rt = 0; rt < 4; ++rt) {
    v8f c0 = {}, c1 = {};
#pragma unroll
    for (int j = 0; j < 2; ++j) {
      v16h a = load_a(tmp, rt * 16, 64, j * 32);
      c0 = wmma32(a, load_b(vT, 0, RAYS, j * 32), c0);
      c1 = wmma32(a, load_b(vT, 16, RAYS, j * 32), c1);
    }
    st_f16(tmp, rt * 16, 0, 64, c0, nullptr, 1.0f);
    st_f16(tmp, rt * 16, 16, 64, c1, nullptr, 1.0f);
  }
#pragma unroll
  for (int rt = 0; rt < 4; ++rt)
#pragma unroll
    for (int ct = 0; ct < 2; ++ct) {
      v8f c = {};
      c = wmma32(load_a(tmp, rt * 16, 64, 0), load_b(w_saWo, ct * 16, EDIM, 0), c);
      acc_f32(x, rt * 16, ct * 16, EDIM, c, p.sa_bo);
    }
  ln_tokens(x, xf, p.sa_ln1_g, p.sa_ln1_b, lane);
  ffn_block(x, xf, tmp, w_saW1, p.sa_b1, w_saW2, p.sa_b2);
  ln_tokens(x, xf, p.sa_ln2_g, p.sa_ln2_b, lane);

  // ---- pool + MLP head (VALU f32; M=1 per sample) ----
  {
    float pe = 0.f;
    for (int t = 0; t < RAYS; ++t) pe += x[t * EDIM + lane];
    s_vec[wave][lane] = pe * (1.0f / RAYS);
  }
#pragma unroll
  for (int jj = 0; jj < 4; ++jj) {
    int j = lane + jj * 32;
    float acc = p.fc_in_b[j];
    for (int k = 0; k < EDIM; ++k)  acc += s_vec[wave][k]  * p.fc_in_W[k * HIDN + j];
    for (int k = 0; k < EDIM; ++k)  acc += s_rich[wave][k] * p.fc_in_W[(EDIM + k) * HIDN + j];
    s_h1[wave][j] = gelu_exact(acc);
  }
#pragma unroll
  for (int jj = 0; jj < 4; ++jj) {
    int j = lane + jj * 32;
    float acc = p.fc1_b[j];
    for (int k = 0; k < HIDN; ++k) acc += s_h1[wave][k] * p.fc1_W[k * HIDN + j];
    s_h2[wave][j] = gelu_exact(acc);
  }
  if (lane < 8) {
    float acc = p.actor_b[lane];
    for (int k = 0; k < HIDN; ++k) acc += s_h2[wave][k] * p.actor_W[k * 8 + lane];
    p.out[(size_t)s * 8 + lane] = acc;
  }
  if (lane == 8) {
    float acc = p.critic_b[0];
    for (int k = 0; k < HIDN; ++k) acc += s_h2[wave][k] * p.critic_W[k];
    p.out[(size_t)p.nB * 8 + s] = acc;
  }
}

extern "C" void kernel_launch(void* const* d_in, const int* in_sizes, int n_in,
                              void* d_out, int out_size, void* d_ws, size_t ws_size,
                              hipStream_t stream) {
  (void)n_in; (void)out_size; (void)d_ws; (void)ws_size;
  Params p;
  int i = 0;
  auto nx = [&]() { return (const float*)d_in[i++]; };
  p.obs = nx();
  p.ray_ln_g = nx(); p.ray_ln_b = nx(); p.ray_W = nx(); p.ray_b = nx();
  p.rich_ln_g = nx(); p.rich_ln_b = nx(); p.rich_W = nx(); p.rich_b = nx();
  p.pos = nx();
  p.ca_Wq = nx(); p.ca_bq = nx(); p.ca_Wk = nx(); p.ca_bk = nx();
  p.ca_Wv = nx(); p.ca_bv = nx(); p.ca_Wo = nx(); p.ca_bo = nx();
  p.ca_ln1_g = nx(); p.ca_ln1_b = nx(); p.ca_W1 = nx(); p.ca_b1 = nx();
  p.ca_W2 = nx(); p.ca_b2 = nx(); p.ca_ln2_g = nx(); p.ca_ln2_b = nx();
  p.sa_Wq = nx(); p.sa_bq = nx(); p.sa_Wk = nx(); p.sa_bk = nx();
  p.sa_Wv = nx(); p.sa_bv = nx(); p.sa_Wo = nx(); p.sa_bo = nx();
  p.sa_ln1_g = nx(); p.sa_ln1_b = nx(); p.sa_W1 = nx(); p.sa_b1 = nx();
  p.sa_W2 = nx(); p.sa_b2 = nx(); p.sa_ln2_g = nx(); p.sa_ln2_b = nx();
  p.fc_in_W = nx(); p.fc_in_b = nx(); p.fc1_W = nx(); p.fc1_b = nx();
  p.actor_W = nx(); p.actor_b = nx(); p.critic_W = nx(); p.critic_b = nx();
  p.out = (float*)d_out;
  p.nB = in_sizes[0] / OBSN;

  int grid = (p.nB + WPB - 1) / WPB;
  brain_fused_kernel<<<grid, WPB * 32, 0, stream>>>(p);
}
